// ElasticViTSelfAttention_80573586473363
// MI455X (gfx1250) — compile-verified, matrix operand order checked by reading, not run on hardware
//
#include <hip/hip_runtime.h>
#include <hip/hip_bf16.h>

// ---------------- problem constants ----------------
#define BB   32
#define SS   577
#define DD   768
#define HH   12
#define DH   64
#define SPAD 608          // S padded to multiple of 32 (19*32)
#define QT   38           // SPAD/16 row tiles
#define KT32 19           // SPAD/32 k-tiles

typedef __attribute__((ext_vector_type(16))) _Float16 v16h;
typedef __attribute__((ext_vector_type(8)))  _Float16 v8h;
typedef __attribute__((ext_vector_type(4)))  _Float16 v4h;
typedef __attribute__((ext_vector_type(8)))  float    v8f;
typedef __attribute__((ext_vector_type(4)))  float    v4f;

__device__ __forceinline__ v16h frag16(const _Float16* p) {
    // CDNA5 16-bit A/B fragment: elements 0..7 at p[0..7], 8..15 at p[16..23]
    v8h lo = *(const v8h*)(p);
    v8h hi = *(const v8h*)(p + 16);
    return __builtin_shufflevector(lo, hi, 0,1,2,3,4,5,6,7,8,9,10,11,12,13,14,15);
}

__device__ __forceinline__ v8f wmma_f16(v16h a, v16h b, v8f c) {
    return __builtin_amdgcn_wmma_f32_16x16x32_f16(false, a, false, b,
                                                  (short)0, c, false, false);
}

__device__ __forceinline__ v4h cvt4(v4f x) {
    v4h r;
    r.x = (_Float16)x.x; r.y = (_Float16)x.y;
    r.z = (_Float16)x.z; r.w = (_Float16)x.w;
    return r;
}

// ---------------------------------------------------------------------------
// Kernel A: fused QKV projection.  grid = (B*QT, 12, 3), block = 128 (4 waves)
// Writes q,k as f16 [B,H,SPAD,DH]; v transposed as f16 [B,H,DH,SPAD].
// ---------------------------------------------------------------------------
__global__ __launch_bounds__(128)
void qkv_proj_kernel(const float* __restrict__ X,
                     const float* __restrict__ Wq, const float* __restrict__ bq,
                     const float* __restrict__ Wk, const float* __restrict__ bk,
                     const float* __restrict__ Wv, const float* __restrict__ bv,
                     _Float16* __restrict__ qb, _Float16* __restrict__ kb,
                     _Float16* __restrict__ vb)
{
    const int bi = blockIdx.x / QT;
    const int rt = blockIdx.x % QT;     // 16-row tile of S
    const int cg = blockIdx.y;          // 64-col group (== head)
    const int ms = blockIdx.z;          // 0=q 1=k 2=v

    const float* W    = (ms == 0) ? Wq : (ms == 1) ? Wk : Wv;
    const float* bias = (ms == 0) ? bq : (ms == 1) ? bk : bv;

    __shared__ alignas(16) _Float16 Xs[16 * DD];     // 24 KB, row-major f16
    __shared__ alignas(16) _Float16 WtT[64 * 32];    //  4 KB, [col][k] transposed

    const int t = threadIdx.x;

    // ---- stage hidden 16x768 tile, fp32 -> f16, vectorized (b128 in, b64 out)
    // 16*768/4 = 3072 float4 chunks, 24 per thread
    #pragma unroll 4
    for (int idx = t; idx < 16 * (DD / 4); idx += 128) {
        const int c4 = idx % (DD / 4);              // float4 column chunk
        const int r  = idx / (DD / 4);              // tile row
        const int grow = rt * 16 + r;
        v4f v = {};
        if (grow < SS)
            v = *(const v4f*)&X[((size_t)bi * SS + grow) * DD + c4 * 4];
        *(v4h*)&Xs[r * DD + c4 * 4] = cvt4(v);
    }

    const int w    = t >> 5;        // wave -> 16-col subtile
    const int L    = t & 31;
    const int half = L >> 4;
    const int lm   = L & 15;
    const int cbase = cg * 64;

    v8f acc = {};
    for (int kk = 0; kk < DD; kk += 32) {
        __syncthreads();
        // ---- stage W[kk..kk+31][cbase..+64] transposed into LDS as f16.
        // Each thread: one col, 4 consecutive k rows -> one b64 LDS store.
        #pragma unroll
        for (int idx = t; idx < 64 * 8; idx += 128) {
            const int col = idx & 63;               // coalesced across lanes
            const int r   = (idx >> 6) * 4;         // k row group
            v4f v;
            v.x = W[(size_t)(kk + r + 0) * DD + cbase + col];
            v.y = W[(size_t)(kk + r + 1) * DD + cbase + col];
            v.z = W[(size_t)(kk + r + 2) * DD + cbase + col];
            v.w = W[(size_t)(kk + r + 3) * DD + cbase + col];
            *(v4h*)&WtT[col * 32 + r] = cvt4(v);
        }
        __syncthreads();

        const _Float16* pa = &Xs[lm * DD + kk + half * 8];
        const _Float16* pb = &WtT[(w * 16 + lm) * 32 + half * 8];
        acc = wmma_f16(frag16(pa), frag16(pb), acc);
    }

    const int col = cbase + w * 16 + lm;     // global output column
    const int h   = col >> 6;
    const int dh  = col & 63;
    const float bval = bias[col];
    const size_t bh  = (size_t)bi * HH + h;

    if (ms == 2) {
        // v transposed: lane's 8 accum elems land on 8 consecutive rows
        // -> one 16B store at vb[(bh*DH+dh)*SPAD + rt*16 + half*8]
        v8h y;
        #pragma unroll
        for (int j = 0; j < 8; ++j) y[j] = (_Float16)(acc[j] + bval);
        *(v8h*)&vb[(bh * DH + dh) * SPAD + rt * 16 + half * 8] = y;
    } else {
        _Float16* dst = (ms == 0) ? qb : kb;
        #pragma unroll
        for (int j = 0; j < 8; ++j) {
            const int row = rt * 16 + j + half * 8;  // always < SPAD
            dst[(bh * SPAD + row) * DH + dh] = (_Float16)(acc[j] + bval);
        }
    }
}

// ---------------------------------------------------------------------------
// Kernel B: attention for one 16-query tile.  grid = (QT, H, B), block = 32.
// ---------------------------------------------------------------------------
__global__ __launch_bounds__(32)
void attn_kernel(const _Float16* __restrict__ qb,
                 const _Float16* __restrict__ kb,
                 const _Float16* __restrict__ vb,
                 float* __restrict__ out)
{
    const int qt = blockIdx.x;
    const int h  = blockIdx.y;
    const int bi = blockIdx.z;

    __shared__ alignas(16) float     sc[16 * SPAD];   // 38.0 KB scores (f32)
    __shared__ alignas(16) _Float16  pr[16 * SPAD];   // 19.0 KB exp probs (f16)
    __shared__ float red[32];
    __shared__ float rowsum[16];

    const int L    = threadIdx.x;
    const int half = L >> 4;
    const int lm   = L & 15;
    const size_t bh = (size_t)bi * HH + h;

    // Q fragments for this tile: K (=dh) 0..31 and 32..63, kept in registers
    const _Float16* qrow = qb + (bh * SPAD + qt * 16 + lm) * DH + half * 8;
    const v16h a0 = frag16(qrow);
    const v16h a1 = frag16(qrow + 32);

    const float scale = 0.125f;   // 1/sqrt(64)

    // ---- scores = q . k^T (38 key tiles of 16), masked + scaled into LDS ----
    for (int nt = 0; nt < QT; ++nt) {
        const int key = nt * 16 + lm;
        const _Float16* krow = kb + (bh * SPAD + key) * DH + half * 8;
        v8f c = {};
        c = wmma_f16(a0, frag16(krow),      c);
        c = wmma_f16(a1, frag16(krow + 32), c);
        const bool valid = key < SS;
        #pragma unroll
        for (int j = 0; j < 8; ++j)
            sc[(j + half * 8) * SPAD + key] = valid ? c[j] * scale : -1.0e30f;
    }
    __syncthreads();

    // ---- softmax: 2 lanes per row, each scans 304 columns ----
    float m = -1.0e30f;
    for (int i = half * 304; i < half * 304 + 304; ++i)
        m = fmaxf(m, sc[lm * SPAD + i]);
    red[L] = m;
    __syncthreads();
    const float rmax = fmaxf(red[lm], red[lm + 16]);
    __syncthreads();

    float sum = 0.0f;
    for (int i = half * 304; i < half * 304 + 304; ++i) {
        float e = __expf(sc[lm * SPAD + i] - rmax);
        pr[lm * SPAD + i] = (_Float16)e;      // unnormalized probs
        sum += e;
    }
    red[L] = sum;
    __syncthreads();
    if (half == 0) rowsum[lm] = red[lm] + red[lm + 16];
    __syncthreads();

    // ---- ctx = P . V  (4 dh-tiles x 19 key-tiles of 32) ----
    for (int dt = 0; dt < 4; ++dt) {
        v8f c = {};
        for (int kt = 0; kt < KT32; ++kt) {
            const _Float16* pp = &pr[lm * SPAD + kt * 32 + half * 8];
            const _Float16* vp = vb + (bh * DH + dt * 16 + lm) * SPAD
                                    + kt * 32 + half * 8;
            c = wmma_f16(frag16(pp), frag16(vp), c);
        }
        #pragma unroll
        for (int j = 0; j < 8; ++j) {
            const int mr = j + half * 8;
            const int s  = qt * 16 + mr;
            if (s < SS)
                out[((size_t)bi * SS + s) * DD + h * 64 + dt * 16 + lm] =
                    c[j] / rowsum[mr];
        }
    }
}

// ---------------------------------------------------------------------------
extern "C" void kernel_launch(void* const* d_in, const int* in_sizes, int n_in,
                              void* d_out, int out_size, void* d_ws, size_t ws_size,
                              hipStream_t stream) {
    const float* X  = (const float*)d_in[0];
    const float* Wq = (const float*)d_in[1];
    const float* bq = (const float*)d_in[2];
    const float* Wk = (const float*)d_in[3];
    const float* bk = (const float*)d_in[4];
    const float* Wv = (const float*)d_in[5];
    const float* bv = (const float*)d_in[6];
    float* out = (float*)d_out;

    const size_t elems = (size_t)BB * HH * SPAD * DH;   // per q/k/v buffer
    _Float16* qb = (_Float16*)d_ws;
    _Float16* kb = qb + elems;
    _Float16* vb = kb + elems;

    qkv_proj_kernel<<<dim3(BB * QT, DD / 64, 3), 128, 0, stream>>>(
        X, Wq, bq, Wk, bk, Wv, bv, qb, kb, vb);

    attn_kernel<<<dim3(QT, HH, BB), 32, 0, stream>>>(qb, kb, vb, out);
}
